// CoattentiveLayer_29532195127848
// MI455X (gfx1250) — compile-verified, matrix operand order checked by reading, not run on hardware
//
#include <hip/hip_runtime.h>

// ---------------------------------------------------------------------------
// CoattentiveLayer for MI455X (gfx1250, wave32, WMMA + async-to-LDS staging)
// B=32, Lc=2048, Lq=512, d=256.  GEMMs: v_wmma_f32_16x16x32_bf16.
// Block tile 128M x 64N (8 waves, each 32x32 / 2x2 WMMA accumulators).
// Both A and B panels are staged into LDS with global_load_async_to_lds_b128,
// double-buffered, synchronized via s_wait_asynccnt + block barriers.
// Padded dims: Lc+1 -> 2080, Lq+1 -> 544.
// ---------------------------------------------------------------------------

typedef __attribute__((ext_vector_type(16))) __bf16 v16bf;
typedef __attribute__((ext_vector_type(8)))  __bf16 v8bf;
typedef __attribute__((ext_vector_type(8)))  float  v8f;

static constexpr int BATCH = 32;
static constexpr int LC    = 2048;
static constexpr int LQ    = 512;
static constexpr int D     = 256;
static constexpr int LC1   = 2049;   // with sentinel
static constexpr int LQ1   = 513;    // with sentinel
static constexpr int LCP   = 2080;   // padded to 65*32
static constexpr int LQP   = 544;    // padded to 17*32
static constexpr float NEG = -1e9f;

static constexpr int KCH   = 64;          // K elems per staged chunk
static constexpr int SROW  = 72;          // LDS row stride (elems): 144B, 16B-aligned,
                                          // 36-bank stride -> 16 rows conflict-free
static constexpr int ABUFE = 128 * SROW;  // elems per A staging buffer
static constexpr int BBUFE = 64 * SROW;   // elems per B staging buffer
static constexpr int SMEME = 2 * ABUFE + 2 * BBUFE;  // 27648 elems = 54KB

// ---------------------------------------------------------------------------
// Fill / convert kernels (fp32 -> bf16, sentinel prepend, zero padding)
// ---------------------------------------------------------------------------

__global__ void k_fill_ctx(const float* __restrict__ context,
                           const float* __restrict__ E,
                           __bf16* __restrict__ ctxb,     // [B][LCP][D]
                           __bf16* __restrict__ ctxTb) {  // [B][D][LCP]
    size_t idx = (size_t)blockIdx.x * blockDim.x + threadIdx.x;
    size_t n = (size_t)BATCH * LCP * D;
    if (idx >= n) return;
    int dd = idx % D;
    int c  = (idx / D) % LCP;
    int b  = idx / ((size_t)D * LCP);
    float v = 0.f;
    if (c == 0)            v = E[dd];
    else if (c <= LC)      v = context[((size_t)b * LC + (c - 1)) * D + dd];
    __bf16 o = (__bf16)v;
    ctxb[((size_t)b * LCP + c) * D + dd]   = o;
    ctxTb[((size_t)b * D + dd) * LCP + c]  = o;
}

__global__ void k_fill_qin(const float* __restrict__ question,
                           const float* __restrict__ E,
                           __bf16* __restrict__ qinb) {   // [B][LQP][D]
    size_t idx = (size_t)blockIdx.x * blockDim.x + threadIdx.x;
    size_t n = (size_t)BATCH * LQP * D;
    if (idx >= n) return;
    int dd = idx % D;
    int q  = (idx / D) % LQP;
    int b  = idx / ((size_t)D * LQP);
    float v = 0.f;
    if (q == 0)            v = E[D + dd];
    else if (q <= LQ)      v = question[((size_t)b * LQ + (q - 1)) * D + dd];
    qinb[((size_t)b * LQP + q) * D + dd] = (__bf16)v;
}

__global__ void k_fill_w(const float* __restrict__ W,
                         __bf16* __restrict__ Wb) {       // [D][D] = W[e][d]
    int idx = blockIdx.x * blockDim.x + threadIdx.x;
    if (idx < D * D) Wb[idx] = (__bf16)W[idx];
}

// ---------------------------------------------------------------------------
// Fragment helpers.  CDNA5 16-bit A/B layout: lane -> (row = lane&15,
// K-half = lane>>4), per-lane halves {K0..7,K16..23} / {K8..15,K24..31}.
// ---------------------------------------------------------------------------

__device__ __forceinline__ v16bf frag16(const __bf16* p) {
    v8bf lo = *(const v8bf*)(p);
    v8bf hi = *(const v8bf*)(p + 16);
    return __builtin_shufflevector(lo, hi, 0,1,2,3,4,5,6,7,8,9,10,11,12,13,14,15);
}

__device__ __forceinline__ void async_ldsB128(unsigned ldsAddr, const __bf16* g) {
    // Async DMA: 16B global -> LDS, tracked by ASYNCcnt.
    asm volatile("global_load_async_to_lds_b128 %0, %1, off"
                 :: "v"(ldsAddr), "v"((unsigned long long)(size_t)g)
                 : "memory");
}

struct Acc22 { v8f a00, a01, a10, a11; };

__device__ __forceinline__ void wmma4(const v16bf& av0, const v16bf& av1,
                                      const v16bf& bv0, const v16bf& bv1,
                                      Acc22& acc) {
    acc.a00 = __builtin_amdgcn_wmma_f32_16x16x32_bf16(false, av0, false, bv0, (short)0, acc.a00, false, false);
    acc.a01 = __builtin_amdgcn_wmma_f32_16x16x32_bf16(false, av0, false, bv1, (short)0, acc.a01, false, false);
    acc.a10 = __builtin_amdgcn_wmma_f32_16x16x32_bf16(false, av1, false, bv0, (short)0, acc.a10, false, false);
    acc.a11 = __builtin_amdgcn_wmma_f32_16x16x32_bf16(false, av1, false, bv1, (short)0, acc.a11, false, false);
}

__device__ __forceinline__ const v8f& accAt(const Acc22& a, int i, int j) {
    return i == 0 ? (j == 0 ? a.a00 : a.a01) : (j == 0 ? a.a10 : a.a11);
}

// ---------------------------------------------------------------------------
// Stage one 64-K chunk of the block's A (128 rows) and B (64 rows) panels
// into LDS.  256 threads x (4 A + 2 B) async b128 ops, fully coalesced
// (2 consecutive threads cover one 128B A row, 4 one B row).  Source rows clamp to
// the last valid row (duplicates are harmless; consumers clamp identically).
// ---------------------------------------------------------------------------

__device__ __forceinline__ void stage_chunk(const __bf16* __restrict__ Ag,
                                            const __bf16* __restrict__ Bg,
                                            int lda, int ldb, int M, int N,
                                            int R0, int C0, int kbase,
                                            unsigned aB, unsigned bB, int tid) {
#pragma unroll
    for (int a = 0; a < 4; ++a) {
        int idx = tid * 4 + a;
        int sr  = idx >> 3;
        int sc  = (idx & 7) * 8;
        int gr  = R0 + sr; if (gr > M - 1) gr = M - 1;
        async_ldsB128(aB + (unsigned)((sr * SROW + sc) * 2),
                      Ag + (size_t)gr * lda + kbase + sc);
    }
#pragma unroll
    for (int a = 0; a < 2; ++a) {
        int idx = tid * 2 + a;
        int sr  = idx >> 3;
        int sc  = (idx & 7) * 8;
        int gc  = C0 + sr; if (gc > N - 1) gc = N - 1;
        async_ldsB128(bB + (unsigned)((sr * SROW + sc) * 2),
                      Bg + (size_t)gc * ldb + kbase + sc);
    }
}

// ---------------------------------------------------------------------------
// Block GEMM mainloop: 128M x 64N block tile, K in double-buffered 64 chunks.
// Per chunk / wave: 16 ds_load_b128 + 8 WMMA; per chunk / thread: 6 asyncs.
// K odd-multiple-of-32 handled by a direct-global 32-wide tail.
// All threads of the block must call this (contains barriers).
// ---------------------------------------------------------------------------

__device__ __forceinline__ Acc22 gemm_block(const __bf16* __restrict__ Ag,
                                            const __bf16* __restrict__ Bg,
                                            int lda, int ldb, int K,
                                            int M, int N, int R0, int C0,
                                            int aSlice, int bSlice,
                                            __bf16* smem, int tid) {
    const int lane = tid & 31;
    const int r    = lane & 15;
    const int hi   = lane >> 4;
    Acc22 acc;
    acc.a00 = {}; acc.a01 = {}; acc.a10 = {}; acc.a11 = {};

    const unsigned sBase = (unsigned)(size_t)smem;     // LDS byte offset
    const unsigned aB0 = sBase;
    const unsigned aB1 = sBase + (unsigned)(ABUFE * 2);
    const unsigned bB0 = sBase + (unsigned)(2 * ABUFE * 2);
    const unsigned bB1 = bB0 + (unsigned)(BBUFE * 2);

    const int nCh = K >> 6;
    stage_chunk(Ag, Bg, lda, ldb, M, N, R0, C0, 0, aB0, bB0, tid);

    for (int ch = 0; ch < nCh; ++ch) {
        if (ch + 1 < nCh) {
            bool odd = ((ch + 1) & 1) != 0;
            stage_chunk(Ag, Bg, lda, ldb, M, N, R0, C0, (ch + 1) * KCH,
                        odd ? aB1 : aB0, odd ? bB1 : bB0, tid);
            asm volatile("s_wait_asynccnt 0x6" ::: "memory");
        } else {
            asm volatile("s_wait_asynccnt 0x0" ::: "memory");
        }
        __syncthreads();   // staged chunk visible to all waves

        const __bf16* la = smem + (ch & 1) * ABUFE;
        const __bf16* lb = smem + 2 * ABUFE + (ch & 1) * BBUFE;
        const __bf16* pa0 = la + (aSlice * 32 + r) * SROW + 8 * hi;
        const __bf16* pa1 = pa0 + 16 * SROW;
        const __bf16* pb0 = lb + (bSlice * 32 + r) * SROW + 8 * hi;
        const __bf16* pb1 = pb0 + 16 * SROW;
#pragma unroll
        for (int s = 0; s < 2; ++s) {
            v16bf av0 = frag16(pa0 + s * 32);
            v16bf av1 = frag16(pa1 + s * 32);
            v16bf bv0 = frag16(pb0 + s * 32);
            v16bf bv1 = frag16(pb1 + s * 32);
            wmma4(av0, av1, bv0, bv1, acc);
        }
        __syncthreads();   // everyone done reading before buffers re-fill
    }

    if (K & 63) {          // 32-wide tail, direct global
        const int kb = nCh * KCH;
        const __bf16* ga0 = Ag + (size_t)(R0 + aSlice * 32 + r) * lda + kb + 8 * hi;
        const __bf16* ga1 = ga0 + (size_t)16 * lda;
        const __bf16* gb0 = Bg + (size_t)(C0 + bSlice * 32 + r) * ldb + kb + 8 * hi;
        const __bf16* gb1 = gb0 + (size_t)16 * ldb;
        wmma4(frag16(ga0), frag16(ga1), frag16(gb0), frag16(gb1), acc);
    }
    return acc;
}

// ---------------------------------------------------------------------------
// GEMM 1: qst = tanh(qin @ W^T + b) -> qstb [B][LQP][D], qstTb [B][D][LQP]
// ---------------------------------------------------------------------------

__global__ void k_gemm_qst(const __bf16* __restrict__ qinb,
                           const __bf16* __restrict__ Wb,
                           const float* __restrict__ bias,
                           __bf16* __restrict__ qstb,
                           __bf16* __restrict__ qstTb) {
    __shared__ __align__(16) __bf16 smem[SMEME];
    const int tid = threadIdx.x, lane = tid & 31, wv = tid >> 5;
    const int mi = wv >> 1, ni = wv & 1;
    constexpr int Mt = LQP / 32, Nt = D / 32;       // 17, 8
    constexpr int gM = (Mt + 3) / 4, gN = (Nt + 1) / 2;  // 5, 4
    const int b   = blockIdx.x / (gN * gM);
    const int rem = blockIdx.x % (gN * gM);
    const int ng  = rem / gM;
    const int mg  = rem % gM;
    int amt = mg * 4 + mi; if (amt > Mt - 1) amt = Mt - 1;
    int bnt = ng * 2 + ni; if (bnt > Nt - 1) bnt = Nt - 1;

    const __bf16* Ag = qinb + (size_t)b * LQP * D;
    const __bf16* Bg = Wb;
    Acc22 acc = gemm_block(Ag, Bg, D, D, D, LQP, D, mg * 128, ng * 64,
                           amt - mg * 4, bnt - ng * 2, smem, tid);

    const int hi = lane >> 4;
#pragma unroll
    for (int j = 0; j < 2; ++j) {
        const int e = bnt * 32 + j * 16 + (lane & 15);
        const float bv = bias[e];
#pragma unroll
        for (int i = 0; i < 2; ++i) {
            const v8f& av = accAt(acc, i, j);
#pragma unroll
            for (int g = 0; g < 8; ++g) {
                int q = amt * 32 + i * 16 + g + 8 * hi;
                float v = tanhf(av[g] + bv);
                __bf16 o = (q < LQ1) ? (__bf16)v : (__bf16)0.f;
                qstb [((size_t)b * LQP + q) * D + e]  = o;
                qstTb[((size_t)b * D + e) * LQP + q]  = o;
            }
        }
    }
}

// ---------------------------------------------------------------------------
// GEMM 2: affinity[b][c][q] = ctx @ qst^T -> fp32 [B][LCP][LQP]
// ---------------------------------------------------------------------------

__global__ void k_gemm_aff(const __bf16* __restrict__ ctxb,
                           const __bf16* __restrict__ qstb,
                           float* __restrict__ aff) {
    __shared__ __align__(16) __bf16 smem[SMEME];
    const int tid = threadIdx.x, lane = tid & 31, wv = tid >> 5;
    const int mi = wv >> 1, ni = wv & 1;
    constexpr int Mt = LCP / 32, Nt = LQP / 32;          // 65, 17
    constexpr int gM = (Mt + 3) / 4, gN = (Nt + 1) / 2;  // 17, 9
    const int b   = blockIdx.x / (gN * gM);
    const int rem = blockIdx.x % (gN * gM);
    const int ng  = rem / gM;
    const int mg  = rem % gM;
    int amt = mg * 4 + mi; if (amt > Mt - 1) amt = Mt - 1;
    int bnt = ng * 2 + ni; if (bnt > Nt - 1) bnt = Nt - 1;

    const __bf16* Ag = ctxb + (size_t)b * LCP * D;
    const __bf16* Bg = qstb + (size_t)b * LQP * D;
    Acc22 acc = gemm_block(Ag, Bg, D, D, D, LCP, LQP, mg * 128, ng * 64,
                           amt - mg * 4, bnt - ng * 2, smem, tid);

    const int hi = lane >> 4;
    float* out = aff + (size_t)b * LCP * LQP;
#pragma unroll
    for (int j = 0; j < 2; ++j) {
        const int n = bnt * 32 + j * 16 + (lane & 15);
#pragma unroll
        for (int i = 0; i < 2; ++i) {
            const v8f& av = accAt(acc, i, j);
#pragma unroll
            for (int g = 0; g < 8; ++g) {
                int m = amt * 32 + i * 16 + g + 8 * hi;
                out[(size_t)m * LQP + n] = av[g];
            }
        }
    }
}

// ---------------------------------------------------------------------------
// Softmax over context (axis c), output transposed: AcT [B][LQP][LCP] bf16.
// ---------------------------------------------------------------------------

__global__ void k_softmax_ctx(const float* __restrict__ aff,
                              const unsigned char* __restrict__ cpad,
                              __bf16* __restrict__ AcT) {
    const int b  = blockIdx.x / (LQP / 32);
    const int qt = blockIdx.x % (LQP / 32);
    const int q0 = qt * 32;
    const int t  = threadIdx.x;
    const int ql = t & 31;     // q within strip (lane)
    const int sl = t >> 5;     // c-slice (wave id), 8 slices
    const int q  = q0 + ql;
    const bool validQ = (q < LQ1);

    const float* affb = aff + (size_t)b * LCP * LQP;
    const unsigned char* cp = cpad + (size_t)b * LC;

    __shared__ float red[8][32];
    __shared__ float tile[32][33];

    float m = -3.0e38f;
    for (int c = sl; c < LC1; c += 8) {
        bool msk = (c > 0) && (cp[c - 1] != 0);
        float v = msk ? NEG : affb[(size_t)c * LQP + q];
        m = fmaxf(m, v);
    }
    red[sl][ql] = m;
    __syncthreads();
    float mq = red[0][ql];
    for (int s = 1; s < 8; ++s) mq = fmaxf(mq, red[s][ql]);
    __syncthreads();

    float s = 0.f;
    for (int c = sl; c < LC1; c += 8) {
        bool msk = (c > 0) && (cp[c - 1] != 0);
        float v = msk ? NEG : affb[(size_t)c * LQP + q];
        s += __expf(v - mq);
    }
    red[sl][ql] = s;
    __syncthreads();
    float sumq = 0.f;
    for (int ss = 0; ss < 8; ++ss) sumq += red[ss][ql];
    float inv = 1.f / sumq;
    __syncthreads();

    for (int c0 = 0; c0 < LCP; c0 += 32) {
        for (int ii = 0; ii < 4; ++ii) {
            int ci = ii * 8 + sl;
            int c  = c0 + ci;
            float w = 0.f;
            if (c < LC1 && validQ) {
                bool msk = (c > 0) && (cp[c - 1] != 0);
                float v = msk ? NEG : affb[(size_t)c * LQP + q];
                w = __expf(v - mq) * inv;
            }
            tile[ci][ql] = w;
        }
        __syncthreads();
        for (int ii = 0; ii < 4; ++ii) {
            int qi = ii * 8 + sl;
            int qq = q0 + qi;
            int c  = c0 + ql;
            AcT[((size_t)b * LQP + qq) * LCP + c] = (__bf16)tile[ql][qi];
        }
        __syncthreads();
    }
}

// ---------------------------------------------------------------------------
// Softmax over question (axis q): AqT [B][LCP][LQP] bf16. One wave per row.
// ---------------------------------------------------------------------------

__global__ void k_softmax_qst(const float* __restrict__ aff,
                              const unsigned char* __restrict__ qpad,
                              __bf16* __restrict__ AqT) {
    const int lane = threadIdx.x & 31;
    const int wv   = threadIdx.x >> 5;
    const int row  = blockIdx.x * 8 + wv;
    const int b = row / LCP;
    const int c = row % LCP;
    const float* rowp = aff + ((size_t)b * LCP + c) * LQP;
    const unsigned char* qp = qpad + (size_t)b * LQ;

    float m = -3.0e38f;
    for (int q = lane; q < LQ1; q += 32) {
        bool msk = (q > 0) && (qp[q - 1] != 0);
        float v = msk ? NEG : rowp[q];
        m = fmaxf(m, v);
    }
    for (int o = 16; o >= 1; o >>= 1) m = fmaxf(m, __shfl_xor(m, o, 32));

    float s = 0.f;
    for (int q = lane; q < LQ1; q += 32) {
        bool msk = (q > 0) && (qp[q - 1] != 0);
        float v = msk ? NEG : rowp[q];
        s += __expf(v - m);
    }
    for (int o = 16; o >= 1; o >>= 1) s += __shfl_xor(s, o, 32);
    float inv = 1.f / s;

    __bf16* dst = AqT + ((size_t)b * LCP + c) * LQP;
    for (int q = lane; q < LQP; q += 32) {
        float w = 0.f;
        if (q < LQ1) {
            bool msk = (q > 0) && (qp[q - 1] != 0);
            float v = msk ? NEG : rowp[q];
            w = __expf(v - m) * inv;
        }
        dst[q] = (__bf16)w;
    }
}

// ---------------------------------------------------------------------------
// GEMM 3: soc = AcT @ ctx  (K = LCP), stored transposed: socTb [B][D][LQP]
// ---------------------------------------------------------------------------

__global__ void k_gemm_soc(const __bf16* __restrict__ AcT,
                           const __bf16* __restrict__ ctxTb,
                           __bf16* __restrict__ socTb) {
    __shared__ __align__(16) __bf16 smem[SMEME];
    const int tid = threadIdx.x, lane = tid & 31, wv = tid >> 5;
    const int mi = wv >> 1, ni = wv & 1;
    constexpr int Mt = LQP / 32, Nt = D / 32;            // 17, 8
    constexpr int gM = (Mt + 3) / 4, gN = (Nt + 1) / 2;  // 5, 4
    const int b   = blockIdx.x / (gN * gM);
    const int rem = blockIdx.x % (gN * gM);
    const int ng  = rem / gM;
    const int mg  = rem % gM;
    int amt = mg * 4 + mi; if (amt > Mt - 1) amt = Mt - 1;
    int bnt = ng * 2 + ni; if (bnt > Nt - 1) bnt = Nt - 1;

    const __bf16* Ag = AcT   + (size_t)b * LQP * LCP;
    const __bf16* Bg = ctxTb + (size_t)b * D * LCP;
    Acc22 acc = gemm_block(Ag, Bg, LCP, LCP, LCP, LQP, D, mg * 128, ng * 64,
                           amt - mg * 4, bnt - ng * 2, smem, tid);

    const int hi = lane >> 4;
#pragma unroll
    for (int j = 0; j < 2; ++j) {
        const int n = bnt * 32 + j * 16 + (lane & 15);
#pragma unroll
        for (int i = 0; i < 2; ++i) {
            const v8f& av = accAt(acc, i, j);
#pragma unroll
            for (int g = 0; g < 8; ++g) {
                int q = amt * 32 + i * 16 + g + 8 * hi;
                socTb[((size_t)b * D + n) * LQP + q] = (__bf16)av[g];
            }
        }
    }
}

// ---------------------------------------------------------------------------
// GEMMs 4/5: out_half = AqT @ Bbuf^T, Bbuf is [B][D][LQP] (qstTb or socTb).
// Writes directly into d_out [B][LC][2*D], dropping sentinel row c=0.
// ---------------------------------------------------------------------------

__global__ void k_gemm_out(const __bf16* __restrict__ AqT,
                           const __bf16* __restrict__ Bbuf,
                           float* __restrict__ out,
                           int colbase) {
    __shared__ __align__(16) __bf16 smem[SMEME];
    const int tid = threadIdx.x, lane = tid & 31, wv = tid >> 5;
    const int mi = wv >> 1, ni = wv & 1;
    constexpr int Mt = LCP / 32, Nt = D / 32;            // 65, 8
    constexpr int gM = (Mt + 3) / 4, gN = (Nt + 1) / 2;  // 17, 4
    const int b   = blockIdx.x / (gN * gM);
    const int rem = blockIdx.x % (gN * gM);
    const int ng  = rem / gM;
    const int mg  = rem % gM;
    int amt = mg * 4 + mi; if (amt > Mt - 1) amt = Mt - 1;
    int bnt = ng * 2 + ni; if (bnt > Nt - 1) bnt = Nt - 1;

    const __bf16* Ag = AqT  + (size_t)b * LCP * LQP;
    const __bf16* Bg = Bbuf + (size_t)b * D * LQP;
    Acc22 acc = gemm_block(Ag, Bg, LQP, LQP, LQP, LCP, D, mg * 128, ng * 64,
                           amt - mg * 4, bnt - ng * 2, smem, tid);

    const int hi = lane >> 4;
#pragma unroll
    for (int j = 0; j < 2; ++j) {
        const int n = bnt * 32 + j * 16 + (lane & 15);
#pragma unroll
        for (int i = 0; i < 2; ++i) {
            const v8f& av = accAt(acc, i, j);
#pragma unroll
            for (int g = 0; g < 8; ++g) {
                int c = amt * 32 + i * 16 + g + 8 * hi;
                if (c >= 1 && c <= LC) {
                    out[((size_t)b * LC + (c - 1)) * (2 * D) + colbase + n] = av[g];
                }
            }
        }
    }
}

// ---------------------------------------------------------------------------
// Host-side launch
// ---------------------------------------------------------------------------

extern "C" void kernel_launch(void* const* d_in, const int* in_sizes, int n_in,
                              void* d_out, int out_size, void* d_ws, size_t ws_size,
                              hipStream_t stream) {
    const float* context          = (const float*)d_in[0];
    const float* question         = (const float*)d_in[1];
    const unsigned char* cpad     = (const unsigned char*)d_in[2];
    const unsigned char* qpad     = (const unsigned char*)d_in[3];
    const float* W                = (const float*)d_in[4];
    const float* bias             = (const float*)d_in[5];
    const float* E                = (const float*)d_in[6];
    float* out                    = (float*)d_out;

    char* ws = (char*)d_ws;
    auto carve = [&](size_t bytes) -> void* {
        void* p = (void*)ws;
        ws += (bytes + 255) & ~(size_t)255;
        return p;
    };
    __bf16* ctxb  = (__bf16*)carve((size_t)BATCH * LCP * D * 2);
    __bf16* ctxTb = (__bf16*)carve((size_t)BATCH * D * LCP * 2);
    __bf16* qinb  = (__bf16*)carve((size_t)BATCH * LQP * D * 2);
    __bf16* Wb    = (__bf16*)carve((size_t)D * D * 2);
    __bf16* qstb  = (__bf16*)carve((size_t)BATCH * LQP * D * 2);
    __bf16* qstTb = (__bf16*)carve((size_t)BATCH * D * LQP * 2);
    float*  aff   = (float*) carve((size_t)BATCH * LCP * LQP * 4);
    __bf16* AcT   = (__bf16*)carve((size_t)BATCH * LQP * LCP * 2);
    __bf16* AqT   = (__bf16*)carve((size_t)BATCH * LCP * LQP * 2);
    __bf16* socTb = (__bf16*)carve((size_t)BATCH * D * LQP * 2);

    const int TB = 256; // 8 waves per block

    {
        size_t n = (size_t)BATCH * LCP * D;
        k_fill_ctx<<<dim3((n + TB - 1) / TB), dim3(TB), 0, stream>>>(context, E, ctxb, ctxTb);
    }
    {
        size_t n = (size_t)BATCH * LQP * D;
        k_fill_qin<<<dim3((n + TB - 1) / TB), dim3(TB), 0, stream>>>(question, E, qinb);
    }
    k_fill_w<<<dim3((D * D + TB - 1) / TB), dim3(TB), 0, stream>>>(W, Wb);

    // GEMM 1: qst = tanh(qin @ W^T + b): blocks = B * gN(4) * gM(5)
    k_gemm_qst<<<dim3(BATCH * 4 * 5), dim3(TB), 0, stream>>>(
        qinb, Wb, bias, qstb, qstTb);

    // GEMM 2: affinity: blocks = B * gN(9) * gM(17)
    k_gemm_aff<<<dim3(BATCH * 9 * 17), dim3(TB), 0, stream>>>(
        ctxb, qstb, aff);

    // softmaxes
    k_softmax_ctx<<<dim3(BATCH * (LQP / 32)), dim3(TB), 0, stream>>>(aff, cpad, AcT);
    k_softmax_qst<<<dim3(BATCH * LCP / 8), dim3(TB), 0, stream>>>(aff, qpad, AqT);

    // GEMM 3: sum_of_context (stored transposed): blocks = B * gN(4) * gM(5)
    k_gemm_soc<<<dim3(BATCH * 4 * 5), dim3(TB), 0, stream>>>(
        AcT, ctxTb, socTb);

    // GEMM 4: coattn_context -> out[:, :, 0:256]: blocks = B * gN(4) * gM(17)
    k_gemm_out<<<dim3(BATCH * 4 * 17), dim3(TB), 0, stream>>>(
        AqT, socTb, out, 0);
    // GEMM 5: sum_of_question -> out[:, :, 256:512]
    k_gemm_out<<<dim3(BATCH * 4 * 17), dim3(TB), 0, stream>>>(
        AqT, qstTb, out, D);
}
